// Attention_61907658604730
// MI455X (gfx1250) — matrix operand order repaired and checked by reading, not compiled, not run on hardware
//
#include <hip/hip_runtime.h>
#include <hip/hip_bf16.h>

// ---------------------------------------------------------------------------
// Types for CDNA5 WMMA (wave32)
// ---------------------------------------------------------------------------
typedef __bf16 bf16_t;
typedef __attribute__((ext_vector_type(16))) __bf16 v16bf;
typedef __attribute__((ext_vector_type(8)))  __bf16 v8bf;
typedef __attribute__((ext_vector_type(8)))  float   v8f;

#define BB_   2
#define TQ_   2048
#define TK_   2048
#define DM_   2048
#define HQ_   32
#define HKV_  8
#define HD_   64
#define KVD_  512   // HKV_*HD_

__device__ __forceinline__ v16bf cat8(v8bf lo, v8bf hi) {
  return __builtin_shufflevector(lo, hi, 0,1,2,3,4,5,6,7,8,9,10,11,12,13,14,15);
}

__device__ __forceinline__ v8f wmma_bf16(v16bf a, v16bf b, v8f c) {
  return __builtin_amdgcn_wmma_f32_16x16x32_bf16(
      /*neg_a=*/false, a, /*neg_b=*/false, b,
      /*c_mod=*/(short)0, c, /*reuse_a=*/false, /*reuse_b=*/false);
}

// A-frag: 16x32 tile of row-major (ld = K) bf16 matrix. Lane l holds row
// (m0 + (l&15)), K values {kb..kb+7, kb+16..kb+23} with kb = k0 + (l>>4)*8.
__device__ __forceinline__ v16bf load_afrag(const bf16_t* base, int ld,
                                            int m0, int k0, int lane) {
  const int m  = m0 + (lane & 15);
  const int kb = k0 + ((lane >> 4) << 3);
  const bf16_t* p = base + (size_t)m * ld + kb;
  v8bf lo = *reinterpret_cast<const v8bf*>(p);
  v8bf hi = *reinterpret_cast<const v8bf*>(p + 16);
  return cat8(lo, hi);
}

// B-frag: 32x16 tile, operand given as Bt = N x K row-major (ld = K).
// Lane l holds column n0 + (l&15), K values {kb..kb+7, kb+16..kb+23} with kb = k0 + (l>>4)*8.
__device__ __forceinline__ v16bf load_bfrag(const bf16_t* base, int ld,
                                            int n0, int k0, int lane) {
  const int n  = n0 + (lane & 15);
  const int kb = k0 + ((lane >> 4) << 3);
  const bf16_t* p = base + (size_t)n * ld + kb;
  v8bf lo = *reinterpret_cast<const v8bf*>(p);
  v8bf hi = *reinterpret_cast<const v8bf*>(p + 16);
  return cat8(lo, hi);
}

// ---------------------------------------------------------------------------
// Elementwise helpers
// ---------------------------------------------------------------------------
__global__ void cvt_f32_bf16(const float* __restrict__ in,
                             bf16_t* __restrict__ out, long n) {
  long i = (long)blockIdx.x * blockDim.x + threadIdx.x;
  if (i < n) out[i] = (bf16_t)in[i];
}

// in: f32 [R][C] row-major  ->  out: bf16 [C][R] row-major (i.e. transposed)
__global__ void transpose_f32_bf16(const float* __restrict__ in,
                                   bf16_t* __restrict__ out, int R, int C) {
  long i = (long)blockIdx.x * blockDim.x + threadIdx.x;
  if (i < (long)R * C) {
    int r = (int)(i / C), c = (int)(i % C);
    out[(size_t)c * R + r] = (bf16_t)in[i];
  }
}

// RoPE: raw bf16 [B*T][H*64] (col = h*64+d) -> out bf16 [B][H][T][64]
__global__ void rope_kernel(const bf16_t* __restrict__ raw,
                            bf16_t* __restrict__ out, int T, int H) {
  long i = (long)blockIdx.x * blockDim.x + threadIdx.x;
  long total = (long)BB_ * T * H * 32;
  if (i >= total) return;
  int d2 = (int)(i & 31);
  int h  = (int)((i >> 5) % H);
  long bt = i / ((long)32 * H);
  int t = (int)(bt % T);
  int b = (int)(bt / T);
  float inv = __powf(10000.0f, -(float)d2 * (1.0f / 32.0f));
  float ang = (float)t * inv;
  float c = __cosf(ang), s = __sinf(ang);
  const bf16_t* row = raw + ((size_t)b * T + t) * ((size_t)H * 64) + h * 64;
  float x1 = (float)row[d2];
  float x2 = (float)row[d2 + 32];
  bf16_t* orow = out + ((((size_t)b * H + h) * T) + t) * 64;
  orow[d2]      = (bf16_t)(x1 * c - x2 * s);
  orow[d2 + 32] = (bf16_t)(x2 * c + x1 * s);
}

// V: raw bf16 [B*TK][512] (col = hk*64+d) -> Vt bf16 [B][8][64][TK]
__global__ void vtrans_kernel(const bf16_t* __restrict__ raw,
                              bf16_t* __restrict__ vt) {
  long i = (long)blockIdx.x * blockDim.x + threadIdx.x;
  long total = (long)BB_ * HKV_ * HD_ * TK_;
  if (i >= total) return;
  int t = (int)(i % TK_);
  long r = i / TK_;
  int d  = (int)(r % HD_);
  int hk = (int)((r / HD_) % HKV_);
  int b  = (int)(r / ((long)HD_ * HKV_));
  vt[i] = raw[((size_t)b * TK_ + t) * KVD_ + hk * 64 + d];
}

// ---------------------------------------------------------------------------
// bf16 WMMA GEMM: C[M][N] = A[M][K] (row-major bf16) x Bt[N][K] (row-major bf16)
// Block: 128 threads = 4 waves (2x2); wave tile 64x32 (4 A-frags x 2 B-frags,
// 8 WMMAs per K=32 step, ~21 FLOP/B of L2 traffic); block tile 128x64.
// Stores f32 (Cf) or bf16 (Cb), whichever pointer is non-null.
// Requires M % 128 == 0, N % 64 == 0, K % 32 == 0.
// ---------------------------------------------------------------------------
__global__ __launch_bounds__(128)
void gemm_bf16_wmma(const bf16_t* __restrict__ A, const bf16_t* __restrict__ Bt,
                    int M, int N, int K,
                    float* __restrict__ Cf, bf16_t* __restrict__ Cb) {
  const int lane = threadIdx.x & 31;
  const int wave = threadIdx.x >> 5;
  const int mw = blockIdx.y * 128 + (wave >> 1) * 64;
  const int nw = blockIdx.x * 64  + (wave & 1) * 32;

  v8f acc[4][2] = {};
  for (int k0 = 0; k0 < K; k0 += 32) {
    // L2 prefetch of the tile after next (gfx1250 global_prefetch_b8 path).
    if (k0 + 64 < K) {
      __builtin_prefetch(A  + (size_t)(mw + (lane & 15)) * K + k0 + 64, 0, 3);
      __builtin_prefetch(Bt + (size_t)(nw + (lane & 15)) * K + k0 + 64, 0, 3);
    }
    v16bf b0 = load_bfrag(Bt, K, nw,      k0, lane);
    v16bf b1 = load_bfrag(Bt, K, nw + 16, k0, lane);
#pragma unroll
    for (int i = 0; i < 4; ++i) {
      v16bf a = load_afrag(A, K, mw + 16 * i, k0, lane);
      acc[i][0] = wmma_bf16(a, b0, acc[i][0]);
      acc[i][1] = wmma_bf16(a, b1, acc[i][1]);
    }
  }

  const int row_off = (lane >> 4) << 3;   // C frag: row = r + 8*(lane>>4)
  const int col_off = lane & 15;
  for (int i = 0; i < 4; ++i)
    for (int j = 0; j < 2; ++j)
      for (int r = 0; r < 8; ++r) {
        size_t row = (size_t)mw + 16 * i + r + row_off;
        size_t col = (size_t)nw + 16 * j + col_off;
        if (Cf) Cf[row * N + col] = acc[i][j][r];
        else    Cb[row * N + col] = (bf16_t)acc[i][j][r];
      }
}

// ---------------------------------------------------------------------------
// Flash attention: one wave (32 threads) per (b, h, 16-row q tile).
// Q: [B][32][TQ][64] bf16 ; K: [B][8][TK][64] bf16 ; Vt: [B][8][64][TK] bf16
// amask: f32 [TQ][TK] ; kpm: u8 [B][TK] ; out: bf16 [B*TQ][2048] (col = h*64+d)
// ---------------------------------------------------------------------------
__global__ __launch_bounds__(32)
void attn_kernel(const bf16_t* __restrict__ Q, const bf16_t* __restrict__ Kk,
                 const bf16_t* __restrict__ Vt, const float* __restrict__ amask,
                 const unsigned char* __restrict__ kpm,
                 bf16_t* __restrict__ out) {
  __shared__ __align__(16) bf16_t pbuf[16 * 32];

  const int lane = threadIdx.x;
  const int q0 = blockIdx.x * 16;
  const int h  = blockIdx.y;
  const int b  = blockIdx.z;
  const int hk = h >> 2;                       // repeat_interleave(4)

  const bf16_t* Qh = Q  + ((size_t)b * HQ_  + h)  * TQ_ * HD_;
  const bf16_t* Kh = Kk + ((size_t)b * HKV_ + hk) * TK_ * HD_;
  const bf16_t* Vh = Vt + ((size_t)b * HKV_ + hk) * HD_ * TK_;

  const v16bf aQ0 = load_afrag(Qh, HD_, q0, 0,  lane);
  const v16bf aQ1 = load_afrag(Qh, HD_, q0, 32, lane);

  const int row_off = (lane >> 4) << 3;
  const int col_off = lane & 15;

  v8f o[4] = {};
  float Mrow[8], Lrow[8];
  for (int r = 0; r < 8; ++r) { Mrow[r] = -1e30f; Lrow[r] = 0.0f; }

  for (int kb = 0; kb < TK_; kb += 32) {
    // ---- scores: two 16x16 C-frags covering keys [kb, kb+32) ----
    v8f s[2] = {};
    for (int half = 0; half < 2; ++half) {
      v16bf bk0 = load_bfrag(Kh, HD_, kb + 16 * half, 0,  lane);
      v16bf bk1 = load_bfrag(Kh, HD_, kb + 16 * half, 32, lane);
      s[half] = wmma_bf16(aQ0, bk0, s[half]);
      s[half] = wmma_bf16(aQ1, bk1, s[half]);
    }
    // ---- scale + attn_mask + key_padding_mask (in C-frag layout) ----
    for (int half = 0; half < 2; ++half) {
      int kc = kb + 16 * half + col_off;
      bool dead = (kpm[(size_t)b * TK_ + kc] != 0);
      for (int r = 0; r < 8; ++r) {
        int qr = q0 + r + row_off;
        float v = s[half][r] * 0.125f + amask[(size_t)qr * TK_ + kc];
        s[half][r] = dead ? -1e30f : v;
      }
    }
    // ---- online softmax (row = fixed per (r, lane-half); cols across 16 lanes)
    for (int r = 0; r < 8; ++r) {
      float mx = fmaxf(s[0][r], s[1][r]);
      for (int m = 1; m < 16; m <<= 1) mx = fmaxf(mx, __shfl_xor(mx, m, 32));
      float mnew = fmaxf(Mrow[r], mx);
      float sc = __expf(Mrow[r] - mnew);
      Mrow[r] = mnew;
      float p0 = __expf(s[0][r] - mnew);
      float p1 = __expf(s[1][r] - mnew);
      s[0][r] = p0; s[1][r] = p1;
      float rs = p0 + p1;
      for (int m = 1; m < 16; m <<= 1) rs += __shfl_xor(rs, m, 32);
      Lrow[r] = Lrow[r] * sc + rs;
      o[0][r] *= sc; o[1][r] *= sc; o[2][r] *= sc; o[3][r] *= sc;
    }
    // ---- relayout P: C-frag -> LDS (row-major 16x32 bf16) -> A-frag ----
    __syncthreads();
    for (int half = 0; half < 2; ++half)
      for (int r = 0; r < 8; ++r)
        pbuf[(r + row_off) * 32 + 16 * half + col_off] = (bf16_t)s[half][r];
    __syncthreads();
    const bf16_t* pp = pbuf + (lane & 15) * 32 + ((lane >> 4) << 3);
    v16bf aP = cat8(*reinterpret_cast<const v8bf*>(pp),
                    *reinterpret_cast<const v8bf*>(pp + 16));
    // ---- O += P(16x32) @ V(32x64): 4 B-frags from Vt ----
    for (int j = 0; j < 4; ++j) {
      v16bf bV = load_bfrag(Vh, TK_, 16 * j, kb, lane);
      o[j] = wmma_bf16(aP, bV, o[j]);
    }
  }

  // ---- normalize and store to [B*TQ][2048] bf16, col = h*64 + d ----
  for (int j = 0; j < 4; ++j)
    for (int r = 0; r < 8; ++r) {
      int qr = q0 + r + row_off;
      int dc = 16 * j + col_off;
      float val = o[j][r] / Lrow[r];
      out[((size_t)b * TQ_ + qr) * DM_ + h * HD_ + dc] = (bf16_t)val;
    }
}

// ---------------------------------------------------------------------------
// Host-side orchestration
// ---------------------------------------------------------------------------
extern "C" void kernel_launch(void* const* d_in, const int* in_sizes, int n_in,
                              void* d_out, int out_size, void* d_ws, size_t ws_size,
                              hipStream_t stream) {
  const float* x_q  = (const float*)d_in[0];
  const float* x_kv = (const float*)d_in[1];
  const float* amask = (const float*)d_in[2];
  const unsigned char* kpm = (const unsigned char*)d_in[3];
  const float* Wq = (const float*)d_in[4];
  const float* Wk = (const float*)d_in[5];
  const float* Wv = (const float*)d_in[6];
  const float* Wo = (const float*)d_in[7];
  float* out = (float*)d_out;

  char* ws = (char*)d_ws;
  size_t off = 0;
  auto alloc = [&](size_t bytes) -> void* {
    off = (off + 255) & ~(size_t)255;
    void* p = ws + off;
    off += bytes;
    return p;
  };

  const long NTQ = (long)BB_ * TQ_;       // 4096 rows
  bf16_t* xq_bf  = (bf16_t*)alloc(NTQ * DM_ * 2);
  bf16_t* xkv_bf = (bf16_t*)alloc(NTQ * DM_ * 2);
  bf16_t* WqT = (bf16_t*)alloc((size_t)DM_ * DM_ * 2);
  bf16_t* WkT = (bf16_t*)alloc((size_t)KVD_ * DM_ * 2);
  bf16_t* WvT = (bf16_t*)alloc((size_t)KVD_ * DM_ * 2);
  bf16_t* WoT = (bf16_t*)alloc((size_t)DM_ * DM_ * 2);
  bf16_t* Qraw = (bf16_t*)alloc(NTQ * DM_ * 2);
  bf16_t* Kraw = (bf16_t*)alloc(NTQ * KVD_ * 2);
  bf16_t* Vraw = (bf16_t*)alloc(NTQ * KVD_ * 2);
  bf16_t* Qbf  = (bf16_t*)alloc(NTQ * DM_ * 2);   // [B][32][TQ][64]
  bf16_t* Kbf  = (bf16_t*)alloc(NTQ * KVD_ * 2);  // [B][8][TK][64]
  bf16_t* Vt   = (bf16_t*)alloc(NTQ * KVD_ * 2);  // [B][8][64][TK]
  bf16_t* attn_bf = (bf16_t*)alloc(NTQ * DM_ * 2);

  const int TPB = 256;
  long n;

  // 1) bf16 staging of activations
  n = NTQ * DM_;
  cvt_f32_bf16<<<(int)((n + TPB - 1) / TPB), TPB, 0, stream>>>(x_q,  xq_bf,  n);
  cvt_f32_bf16<<<(int)((n + TPB - 1) / TPB), TPB, 0, stream>>>(x_kv, xkv_bf, n);

  // 2) transposed bf16 weights (Wt = N x K row-major)
  n = (long)DM_ * DM_;
  transpose_f32_bf16<<<(int)((n + TPB - 1) / TPB), TPB, 0, stream>>>(Wq, WqT, DM_, DM_);
  transpose_f32_bf16<<<(int)((n + TPB - 1) / TPB), TPB, 0, stream>>>(Wo, WoT, DM_, DM_);
  n = (long)DM_ * KVD_;
  transpose_f32_bf16<<<(int)((n + TPB - 1) / TPB), TPB, 0, stream>>>(Wk, WkT, DM_, KVD_);
  transpose_f32_bf16<<<(int)((n + TPB - 1) / TPB), TPB, 0, stream>>>(Wv, WvT, DM_, KVD_);

  // 3) projections (bf16 WMMA, bf16 outputs); block tile 128x64
  gemm_bf16_wmma<<<dim3(DM_ / 64,  (int)(NTQ / 128)), 128, 0, stream>>>(
      xq_bf,  WqT, (int)NTQ, DM_,  DM_, nullptr, Qraw);
  gemm_bf16_wmma<<<dim3(KVD_ / 64, (int)(NTQ / 128)), 128, 0, stream>>>(
      xkv_bf, WkT, (int)NTQ, KVD_, DM_, nullptr, Kraw);
  gemm_bf16_wmma<<<dim3(KVD_ / 64, (int)(NTQ / 128)), 128, 0, stream>>>(
      xkv_bf, WvT, (int)NTQ, KVD_, DM_, nullptr, Vraw);

  // 4) RoPE + V transpose into attention-friendly layouts
  n = (long)BB_ * TQ_ * HQ_ * 32;
  rope_kernel<<<(int)((n + TPB - 1) / TPB), TPB, 0, stream>>>(Qraw, Qbf, TQ_, HQ_);
  n = (long)BB_ * TK_ * HKV_ * 32;
  rope_kernel<<<(int)((n + TPB - 1) / TPB), TPB, 0, stream>>>(Kraw, Kbf, TK_, HKV_);
  n = (long)BB_ * HKV_ * HD_ * TK_;
  vtrans_kernel<<<(int)((n + TPB - 1) / TPB), TPB, 0, stream>>>(Vraw, Vt);

  // 5) flash attention (WMMA + online softmax), output in [B*TQ][2048] bf16
  attn_kernel<<<dim3(TQ_ / 16, HQ_, BB_), 32, 0, stream>>>(
      Qbf, Kbf, Vt, amask, kpm, attn_bf);

  // 6) output projection, f32 result straight into d_out
  gemm_bf16_wmma<<<dim3(DM_ / 64, (int)(NTQ / 128)), 128, 0, stream>>>(
      attn_bf, WoT, (int)NTQ, DM_, DM_, out, nullptr);
}
